// output_layer_60438779789508
// MI455X (gfx1250) — matrix operand order repaired and checked by reading, not compiled, not run on hardware
//
#include <hip/hip_runtime.h>

// ---- problem constants (match reference) ----
#define BATCH    256
#define FEAT     1024
#define TIME     500
#define UNITS    128
#define ALPHA    0.9f
#define BETA     0.8f

// ---- tiling ----
#define TG       64                    // time steps per register-blocked group (4 WMMA M-tiles)
#define NG       8                     // ceil(512/64) groups (last: 52 valid t)
#define FCH      64                    // K chunk staged in LDS per round (2 bf16 WMMA k-steps)
#define NFCH     (FEAT / FCH)          // 16
#define NKCG     (FEAT / 32)           // 32 global K=32 chunks

typedef float  v8f   __attribute__((ext_vector_type(8)));
typedef __bf16 v16bf __attribute__((ext_vector_type(16)));
typedef __bf16 bf2   __attribute__((ext_vector_type(2)));

// LDS (bytes): Ah[4][2][32][16]bf16 | Al same | H[8][TG][16]f32 | Mb[128]f32
#define SM_AH_B   0
#define SM_AL_B   8192
#define SM_H_B    16384
#define SM_M_B    (16384 + 8 * TG * 16 * 4)    // 16384 + 32768 = 49152
#define SM_BYTES  (SM_M_B + UNITS * 4)         // 49664 < 64KB

__device__ __forceinline__ void bsplit(float x, __bf16& h, __bf16& l) {
    h = (__bf16)x;
    l = (__bf16)(x - (float)h);
}

// ---- pre-kernel: split W (F,U) f32 -> hi/lo bf16 in WMMA B-fragment order ----
// layout: [kcg 0..31][ut 0..7][lane 0..31][s 0..15], lane = (u&15) + 16*((k>>4)&1), s = k&15
__global__ __launch_bounds__(256)
void wsplit_60438779789508_kernel(const float* __restrict__ w,
                                  __bf16* __restrict__ Wh, __bf16* __restrict__ Wl) {
    const int e = blockIdx.x * 256 + threadIdx.x;      // 0 .. F*U-1 (grid=512)
    const int k  = e >> 7;                             // feature (K)
    const int n  = e & 127;                            // unit (N)
    float x = w[e];
    __bf16 h, l; bsplit(x, h, l);
    const int kcg  = k >> 5, kk = k & 31;
    const int ut   = n >> 4, nc = n & 15;
    const int lane = nc + 16 * (kk >> 4);
    const int s    = kk & 15;
    const size_t idx = (((size_t)(kcg * 8 + ut)) * 32 + lane) * 16 + s;
    Wh[idx] = h; Wl[idx] = l;
}

__global__ __launch_bounds__(256)
void output_layer_60438779789508_kernel(const float* __restrict__ x,     // (B,F,T) f32
                                        const __bf16* __restrict__ Whg,  // swizzled B-frags hi
                                        const __bf16* __restrict__ Wlg,  // swizzled B-frags lo
                                        float* __restrict__ out) {       // (B,U) f32
    const int b    = blockIdx.x;
    const int tid  = threadIdx.x;
    const int lane = tid & 31;
    const int wave = tid >> 5;          // 0..7 == u-tile
    const int u0   = wave * 16;

    extern __shared__ char smem[];
    __bf16* AhL = (__bf16*)(smem + SM_AH_B);           // [q 0..3][kc 0..1][lane][16]
    __bf16* AlL = (__bf16*)(smem + SM_AL_B);
    float*  Hw  = (float*)(smem + SM_H_B) + wave * (TG * 16);
    float*  Mb  = (float*)(smem + SM_M_B);

    const float* xb = x + (size_t)b * FEAT * TIME;

    // scan state: lanes 0..15 own chain u = u0 + lane
    float vmem = 0.0f, syn = 0.0f, mx = -__builtin_inff();

    const int thalf = (lane >> 4) * 8;                 // C-tile row offset for this lane half

    for (int g = 0; g < NG; ++g) {
        const int t0g = g * TG;
        v8f acc[4];
        #pragma unroll
        for (int q = 0; q < 4; ++q) acc[q] = (v8f){0.f,0.f,0.f,0.f,0.f,0.f,0.f,0.f};

        for (int fc = 0; fc < NFCH; ++fc) {
            const int f0 = fc * FCH;
            __syncthreads();   // previous A chunk fully consumed by all waves

            // ---- stage A chunk (64 f x 64 t), split to bf16 hi/lo, pre-swizzled frags.
            // 2048 (k,k+1) pairs, 8 per thread; consecutive tid -> consecutive t (coalesced).
            #pragma unroll
            for (int i = 0; i < 8; ++i) {
                const int p     = tid + 256 * i;
                const int m     = p & 63;              // t within group
                const int k     = (p >> 6) * 2;        // even f_local 0..62
                int tg = t0g + m; if (tg > TIME - 1) tg = TIME - 1;   // clamp (tail unused)
                const float x0 = xb[(size_t)(f0 + k)     * TIME + tg];
                const float x1 = xb[(size_t)(f0 + k + 1) * TIME + tg];
                __bf16 h0, l0, h1, l1; bsplit(x0, h0, l0); bsplit(x1, h1, l1);
                const int q    = m >> 4;               // WMMA tile within group
                const int kc   = k >> 5;               // k-step within chunk
                const int kkk  = k & 31;
                const int ln   = (m & 15) + 16 * (kkk >> 4);
                const int v    = (kkk & 15) >> 1;
                const int base = (((q * 2 + kc) * 32 + ln) * 16) + v * 2;  // even -> 4B aligned
                *(bf2*)(AhL + base) = (bf2){h0, h1};
                *(bf2*)(AlL + base) = (bf2){l0, l1};
            }
            __syncthreads();

            // ---- 2 k-steps of K=32: bf16x3 emulated-f32 WMMA, 4 time-tiles each
            #pragma unroll
            for (int kc = 0; kc < 2; ++kc) {
                const int kcg = fc * 2 + kc;
                const size_t bidx = (((size_t)(kcg * 8 + wave)) * 32 + lane) * 16;
                const v16bf bh = *(const v16bf*)(Whg + bidx);   // 2x global_load_b128 (L2-hot)
                const v16bf bl = *(const v16bf*)(Wlg + bidx);
                #pragma unroll
                for (int q = 0; q < 4; ++q) {
                    const int abase = ((q * 2 + kc) * 32 + lane) * 16;
                    const v16bf ah = *(const v16bf*)(AhL + abase);  // ds_load_b128 x2
                    const v16bf al = *(const v16bf*)(AlL + abase);
                    acc[q] = __builtin_amdgcn_wmma_f32_16x16x32_bf16(
                                 false, ah, false, bl, (short)0, acc[q], false, false);
                    acc[q] = __builtin_amdgcn_wmma_f32_16x16x32_bf16(
                                 false, al, false, bh, (short)0, acc[q], false, false);
                    acc[q] = __builtin_amdgcn_wmma_f32_16x16x32_bf16(
                                 false, ah, false, bh, (short)0, acc[q], false, false);
                }
            }
        }

        // ---- spill H group (64 t x 16 u) to this wave's LDS region
        #pragma unroll
        for (int q = 0; q < 4; ++q)
            #pragma unroll
            for (int r = 0; r < 8; ++r)
                Hw[(q * 16 + thalf + r) * 16 + (lane & 15)] = acc[q][r];
        __syncthreads();   // cross-lane-half visibility + compiler fence

        // ---- sequential synapse/membrane recurrence over 64 t
        if (lane < 16) {
            #pragma unroll
            for (int tt = 0; tt < TG; ++tt) {
                if (t0g + tt < TIME) {
                    const float h = Hw[tt * 16 + lane];
                    syn  = ALPHA * syn + h;       // i_t
                    vmem = BETA * vmem + syn;     // v_t
                    mx   = fmaxf(mx, vmem);
                }
            }
        }
    }

    // ---- fused log_softmax over UNITS for this b
    if (lane < 16) Mb[u0 + lane] = mx;
    __syncthreads();
    if (tid < UNITS) {
        float M = -__builtin_inff();
        for (int u = 0; u < UNITS; ++u) M = fmaxf(M, Mb[u]);
        float S = 0.0f;
        for (int u = 0; u < UNITS; ++u) S += expf(Mb[u] - M);
        out[(size_t)b * UNITS + tid] = Mb[tid] - M - logf(S);
    }
}

extern "C" void kernel_launch(void* const* d_in, const int* in_sizes, int n_in,
                              void* d_out, int out_size, void* d_ws, size_t ws_size,
                              hipStream_t stream) {
    const float* x = (const float*)d_in[0];   // (256,1024,500) f32
    const float* w = (const float*)d_in[1];   // (1024,128)     f32
    float* out = (float*)d_out;               // (256,128)      f32

    // workspace: swizzled bf16 hi/lo W fragments (2 x 256 KB)
    __bf16* Wh = (__bf16*)d_ws;
    __bf16* Wl = Wh + (size_t)FEAT * UNITS;

    wsplit_60438779789508_kernel<<<dim3((FEAT * UNITS) / 256), dim3(256), 0, stream>>>(w, Wh, Wl);
    output_layer_60438779789508_kernel<<<dim3(BATCH), dim3(256), SM_BYTES, stream>>>(x, Wh, Wl, out);
}
